// LstmFromScratchWithWC_24283745092031
// MI455X (gfx1250) — matrix operand order repaired and checked by reading, not compiled, not run on hardware
//
#include <hip/hip_runtime.h>
#include <hip/hip_bf16.h>
#include <math.h>

// ---------------- problem dims ----------------
#define Bsz 128
#define Tsz 1024
#define Isz 512
#define Hsz 1024
#define G4  (4 * Hsz)   // 4096
#define G3  (3 * Hsz)   // 3072  (i,f,g gate columns)
#define P2  (2 * Hsz)   // 2048  (peep columns)

// ---------------- launch config ----------------
#define NWG    64
#define BLOCK  256
#define NWAVES (NWG * (BLOCK / 32))   // 512 wave32s
#define GTH    (NWG * BLOCK)          // 16384 threads

// tiles: gates 8x192, peep 8x128 (16x16 tiles) -> 2560 tiles, 5 per wave
#define NT_GCOL (G3 / 16)             // 192
#define NT_PCOL (P2 / 16)             // 128
#define NT_GATE (8 * NT_GCOL)         // 1536
#define NT_TOTAL (NT_GATE + 8 * NT_PCOL) // 2560

typedef __attribute__((ext_vector_type(16))) __bf16 v16bf;
typedef __attribute__((ext_vector_type(8)))  float  v8f;

union V16U { uint4 q[2]; v16bf v; };

__device__ __forceinline__ unsigned short f32_to_bf16(float f) {
    unsigned u = __float_as_uint(f);
    u += 0x7FFFu + ((u >> 16) & 1u);      // round-to-nearest-even
    return (unsigned short)(u >> 16);
}

__device__ __forceinline__ float sigmoidf_(float x) {
    return 1.0f / (1.0f + __expf(-x));
}

// A-tile (16x32 bf16, row-major src, ld multiple of 8):
// lanes 0-15: row=lane,    K = {0..7, 16..23}
// lanes 16-31: row=lane&15, K = {8..15, 24..31}
__device__ __forceinline__ v16bf load_tile_A(const unsigned short* base, int ld, int lane) {
    const unsigned short* p = base + (size_t)(lane & 15) * ld + ((lane >> 4) << 3);
    V16U u;
    u.q[0] = *reinterpret_cast<const uint4*>(p);        // K chunk +0..7
    u.q[1] = *reinterpret_cast<const uint4*>(p + 16);   // K chunk +16..23
    return u.v;
}

// B-tile (32x16 bf16) from N-major (transposed) weights, row-major [N][K]:
// lanes 0-15: col=lane, K = {0..7, 16..23}; lanes 16-31: col=lane&15, K = {8..15, 24..31}
__device__ __forceinline__ v16bf load_tile_B(const unsigned short* base, int ld, int lane) {
    const unsigned short* p = base + (size_t)(lane & 15) * ld + ((lane >> 4) << 3);
    V16U u;
    u.q[0] = *reinterpret_cast<const uint4*>(p);
    u.q[1] = *reinterpret_cast<const uint4*>(p + 16);
    return u.v;
}

__device__ __forceinline__ void grid_barrier(unsigned* counter, unsigned target) {
    __syncthreads();
    if (threadIdx.x == 0) {
        __hip_atomic_fetch_add(counter, 1u, __ATOMIC_RELEASE, __HIP_MEMORY_SCOPE_AGENT);
        while (__hip_atomic_load(counter, __ATOMIC_ACQUIRE, __HIP_MEMORY_SCOPE_AGENT) < target) {
            __builtin_amdgcn_s_sleep(2);
        }
    }
    __syncthreads();
}

// ---------------- prep: transpose+convert weights to bf16, zero state ----------------
__global__ void lstm_prep_kernel(const float* __restrict__ W, const float* __restrict__ U,
                                 const float* __restrict__ P,
                                 unsigned short* __restrict__ Wt, unsigned short* __restrict__ Ut,
                                 unsigned short* __restrict__ Pt,
                                 unsigned short* __restrict__ c16, float* __restrict__ c32,
                                 unsigned* __restrict__ counter) {
    long tid = (long)blockIdx.x * blockDim.x + threadIdx.x;
    long stride = (long)gridDim.x * blockDim.x;
    if (tid == 0) *counter = 0u;
    for (long i = tid; i < (long)G3 * Isz; i += stride) {      // Wt[n][k] = W[k][n], n<3H
        long n = i / Isz, k = i % Isz;
        Wt[i] = f32_to_bf16(W[k * G4 + n]);
    }
    for (long i = tid; i < (long)G3 * Hsz; i += stride) {      // Ut[n][k] = U[k][n], n<3H
        long n = i / Hsz, k = i % Hsz;
        Ut[i] = f32_to_bf16(U[k * G4 + n]);
    }
    for (long i = tid; i < (long)P2 * Hsz; i += stride) {      // Pt[n][k] = P[k][n]
        long n = i / Hsz, k = i % Hsz;
        Pt[i] = f32_to_bf16(P[k * P2 + n]);
    }
    for (long i = tid; i < (long)Bsz * Hsz; i += stride) {     // c0 = 0
        c32[i] = 0.0f;
        c16[i] = 0;
    }
}

// ---------------- persistent recurrence kernel ----------------
__global__ __launch_bounds__(BLOCK) void lstm_persistent_kernel(
    const float* __restrict__ x, const int* __restrict__ lens,
    const float* __restrict__ W, const float* __restrict__ U,
    const float* __restrict__ P_o, const float* __restrict__ bias,
    const unsigned short* __restrict__ Wt, const unsigned short* __restrict__ Ut,
    const unsigned short* __restrict__ Pt,
    unsigned short* __restrict__ xt16, unsigned short* __restrict__ c16,
    float* __restrict__ c32, float* __restrict__ gates, float* __restrict__ peep,
    float* __restrict__ og, unsigned* __restrict__ counter, float* __restrict__ out) {

    const int wave = blockIdx.x * (BLOCK / 32) + (threadIdx.x >> 5);
    const int lane = threadIdx.x & 31;
    const int gtid = blockIdx.x * BLOCK + threadIdx.x;
    unsigned target = 0;

    // ---- prologue: convert x_0 -> bf16 ----
    for (int i = gtid; i < Bsz * Isz; i += GTH) {
        int b = i >> 9, k = i & (Isz - 1);
        xt16[i] = f32_to_bf16(x[(size_t)b * Tsz * Isz + k]);
    }
    target += NWG; grid_barrier(counter, target);

    for (int t = 0; t < Tsz; ++t) {
        // ---- Phase B: WMMA GEMMs: gates(3H) = x_t@W + c@U ; peep(2H) = c@P ----
        for (int tile = wave; tile < NT_TOTAL; tile += NWAVES) {
            v8f acc = {};
            if (tile < NT_GATE) {
                int rt = tile / NT_GCOL, ct = tile % NT_GCOL;
                const unsigned short* Ax = xt16 + (size_t)rt * 16 * Isz;
                const unsigned short* Bx = Wt + (size_t)ct * 16 * Isz;
                for (int k = 0; k < Isz; k += 32) {
                    v16bf a = load_tile_A(Ax + k, Isz, lane);
                    v16bf b = load_tile_B(Bx + k, Isz, lane);
                    acc = __builtin_amdgcn_wmma_f32_16x16x32_bf16(false, a, false, b,
                                                                  (short)0, acc, false, false);
                }
                const unsigned short* Ac = c16 + (size_t)rt * 16 * Hsz;
                const unsigned short* Bc = Ut + (size_t)ct * 16 * Hsz;
                for (int k = 0; k < Hsz; k += 32) {
                    v16bf a = load_tile_A(Ac + k, Hsz, lane);
                    v16bf b = load_tile_B(Bc + k, Hsz, lane);
                    acc = __builtin_amdgcn_wmma_f32_16x16x32_bf16(false, a, false, b,
                                                                  (short)0, acc, false, false);
                }
                float* op = gates + ((size_t)(rt * 16 + ((lane >> 4) << 3))) * G3 + ct * 16 + (lane & 15);
                #pragma unroll
                for (int j = 0; j < 8; ++j) op[(size_t)j * G3] = acc[j];
            } else {
                int p = tile - NT_GATE;
                int rt = p / NT_PCOL, ct = p % NT_PCOL;
                const unsigned short* Ac = c16 + (size_t)rt * 16 * Hsz;
                const unsigned short* Bc = Pt + (size_t)ct * 16 * Hsz;
                for (int k = 0; k < Hsz; k += 32) {
                    v16bf a = load_tile_A(Ac + k, Hsz, lane);
                    v16bf b = load_tile_B(Bc + k, Hsz, lane);
                    acc = __builtin_amdgcn_wmma_f32_16x16x32_bf16(false, a, false, b,
                                                                  (short)0, acc, false, false);
                }
                float* op = peep + ((size_t)(rt * 16 + ((lane >> 4) << 3))) * P2 + ct * 16 + (lane & 15);
                #pragma unroll
                for (int j = 0; j < 8; ++j) op[(size_t)j * P2] = acc[j];
            }
        }
        // o-gate GEMV (fp32, uses pre-update c) for batches that end at this t
        for (int b = 0; b < Bsz; ++b) {
            if (lens[b] - 1 != t) continue;
            const float* xr = x + ((size_t)b * Tsz + t) * Isz;
            const float* cr = c32 + (size_t)b * Hsz;
            for (int h = gtid; h < Hsz; h += GTH) {
                float s = bias[3 * Hsz + h];
                for (int k = 0; k < Isz; ++k) s += xr[k] * W[(size_t)k * G4 + 3 * Hsz + h];
                for (int k = 0; k < Hsz; ++k) s += cr[k] * U[(size_t)k * G4 + 3 * Hsz + h];
                og[b * Hsz + h] = s;
            }
        }
        // emit h outputs for step t-1 (og from B(t-1), c_new from C(t-1))
        if (t > 0) {
            int ts = t - 1;
            for (int b = 0; b < Bsz; ++b) {
                if (lens[b] - 1 != ts) continue;
                const float* cr = c32 + (size_t)b * Hsz;
                for (int h = gtid; h < Hsz; h += GTH) {
                    float po = 0.0f;
                    for (int k = 0; k < Hsz; ++k) po += cr[k] * P_o[(size_t)k * Hsz + h];
                    float o = sigmoidf_(og[b * Hsz + h] + tanhf(po));
                    out[b * Hsz + h] = o * tanhf(cr[h]);
                }
            }
        }
        target += NWG; grid_barrier(counter, target);

        // ---- Phase C: elementwise cell update + convert x_{t+1} ----
        for (int i = gtid; i < Bsz * Hsz; i += GTH) {
            int b = i >> 10, h = i & (Hsz - 1);
            float gi = gates[(size_t)b * G3 + h]           + bias[h];
            float gf = gates[(size_t)b * G3 + Hsz + h]     + bias[Hsz + h];
            float gg = gates[(size_t)b * G3 + 2 * Hsz + h] + bias[2 * Hsz + h];
            float pi = peep[(size_t)b * P2 + h];
            float pf = peep[(size_t)b * P2 + Hsz + h];
            float it = sigmoidf_(gi + tanhf(pi));
            float ft = sigmoidf_(gf + tanhf(pf));
            float gt = tanhf(gg);
            float cn = ft * c32[i] + it * gt;
            c32[i] = cn;
            c16[i] = f32_to_bf16(cn);
        }
        if (t + 1 < Tsz) {   // safe: all GEMM reads of xt16 completed before last barrier
            for (int i = gtid; i < Bsz * Isz; i += GTH) {
                int b = i >> 9, k = i & (Isz - 1);
                xt16[i] = f32_to_bf16(x[((size_t)b * Tsz + (t + 1)) * Isz + k]);
            }
        }
        target += NWG; grid_barrier(counter, target);
    }

    // ---- tail: outputs for batches ending at t = T-1 ----
    {
        int ts = Tsz - 1;
        for (int b = 0; b < Bsz; ++b) {
            if (lens[b] - 1 != ts) continue;
            const float* cr = c32 + (size_t)b * Hsz;
            for (int h = gtid; h < Hsz; h += GTH) {
                float po = 0.0f;
                for (int k = 0; k < Hsz; ++k) po += cr[k] * P_o[(size_t)k * Hsz + h];
                float o = sigmoidf_(og[b * Hsz + h] + tanhf(po));
                out[b * Hsz + h] = o * tanhf(cr[h]);
            }
        }
    }
}

extern "C" void kernel_launch(void* const* d_in, const int* in_sizes, int n_in,
                              void* d_out, int out_size, void* d_ws, size_t ws_size,
                              hipStream_t stream) {
    const float* x    = (const float*)d_in[0];
    const int*   lens = (const int*)d_in[1];
    const float* W    = (const float*)d_in[2];
    const float* U    = (const float*)d_in[3];
    const float* P    = (const float*)d_in[4];
    const float* P_o  = (const float*)d_in[5];
    const float* bias = (const float*)d_in[6];
    float* out = (float*)d_out;
    (void)in_sizes; (void)n_in; (void)out_size; (void)ws_size;

    char* ws = (char*)d_ws;
    size_t off = 0;
    auto alloc = [&](size_t bytes) -> char* {
        char* p = ws + off;
        off += (bytes + 255) & ~(size_t)255;
        return p;
    };
    unsigned*       counter = (unsigned*)alloc(sizeof(unsigned));
    unsigned short* Wt   = (unsigned short*)alloc((size_t)G3 * Isz * 2);   // 3 MB
    unsigned short* Ut   = (unsigned short*)alloc((size_t)G3 * Hsz * 2);   // 6 MB
    unsigned short* Pt   = (unsigned short*)alloc((size_t)P2 * Hsz * 2);   // 4 MB
    unsigned short* xt16 = (unsigned short*)alloc((size_t)Bsz * Isz * 2);  // 128 KB
    unsigned short* c16  = (unsigned short*)alloc((size_t)Bsz * Hsz * 2);  // 256 KB
    float*          c32  = (float*)alloc((size_t)Bsz * Hsz * 4);           // 512 KB
    float*          gates= (float*)alloc((size_t)Bsz * G3 * 4);            // 1.5 MB
    float*          peep = (float*)alloc((size_t)Bsz * P2 * 4);            // 1 MB
    float*          og   = (float*)alloc((size_t)Bsz * Hsz * 4);           // 512 KB

    hipLaunchKernelGGL(lstm_prep_kernel, dim3(2048), dim3(256), 0, stream,
                       W, U, P, Wt, Ut, Pt, c16, c32, counter);
    hipLaunchKernelGGL(lstm_persistent_kernel, dim3(NWG), dim3(BLOCK), 0, stream,
                       x, lens, W, U, P_o, bias, Wt, Ut, Pt,
                       xt16, c16, c32, gates, peep, og, counter, out);
}
